// SimpleAttention_80221399155305
// MI455X (gfx1250) — matrix operand order repaired and checked by reading, not compiled, not run on hardware
//
#include <hip/hip_runtime.h>
#include <hip/hip_bf16.h>

#define BATCH 4
#define SEQ   2048
#define DMODEL 192
#define NHEAD 4
#define DHEAD 48
#define DPAD  64
#define NEG_INF_F (-1000000000.0f)

typedef __attribute__((ext_vector_type(16))) _Float16 v16h;
typedef __attribute__((ext_vector_type(8)))  _Float16 v8h;
typedef __attribute__((ext_vector_type(8)))  float    v8f;
typedef __attribute__((ext_vector_type(4)))  float    v4f;

__device__ inline v8f zero8() {
  v8f z;
#pragma unroll
  for (int i = 0; i < 8; ++i) z[i] = 0.0f;
  return z;
}

// A-fragment (16x32 f16, M=lane&15, K contiguous: 16 elements from base) from f32 memory.
__device__ inline v16h a_frag_f32(const float* __restrict__ p) {
  v4f x0 = *(const v4f*)p,        x1 = *(const v4f*)(p + 4);
  v4f y0 = *(const v4f*)(p + 8), y1 = *(const v4f*)(p + 12);
  v16h a;
#pragma unroll
  for (int i = 0; i < 4; ++i) {
    a[i]      = (_Float16)x0[i];
    a[4 + i]  = (_Float16)x1[i];
    a[8 + i]  = (_Float16)y0[i];
    a[12 + i] = (_Float16)y1[i];
  }
  return a;
}

// B-fragment (32x16 f16, N=lane&15, K contiguous: (lane>>4)*16 + e) from f32.
__device__ inline v16h b_frag_f32(const float* __restrict__ p) {
  v4f x0 = *(const v4f*)p,       x1 = *(const v4f*)(p + 4);
  v4f x2 = *(const v4f*)(p + 8), x3 = *(const v4f*)(p + 12);
  v16h b;
#pragma unroll
  for (int i = 0; i < 4; ++i) {
    b[i]      = (_Float16)x0[i];
    b[4 + i]  = (_Float16)x1[i];
    b[8 + i]  = (_Float16)x2[i];
    b[12 + i] = (_Float16)x3[i];
  }
  return b;
}

// A-fragment from f16 memory: 16 contiguous halves at p.
__device__ inline v16h a_frag_f16(const _Float16* __restrict__ p) {
  v8h lo = *(const v8h*)p;
  v8h hi = *(const v8h*)(p + 8);
  v16h a;
#pragma unroll
  for (int i = 0; i < 8; ++i) { a[i] = lo[i]; a[8 + i] = hi[i]; }
  return a;
}

// B-fragment from f16 memory: 16 contiguous halves at p.
__device__ inline v16h b_frag_f16(const _Float16* __restrict__ p) {
  v8h lo = *(const v8h*)p;
  v8h hi = *(const v8h*)(p + 8);
  v16h b;
#pragma unroll
  for (int i = 0; i < 8; ++i) { b[i] = lo[i]; b[8 + i] = hi[i]; }
  return b;
}

// Reductions across the 16-lane halves of a wave32 (rows of the C fragment).
__device__ inline float hmax16(float x) {
#pragma unroll
  for (int m = 1; m < 16; m <<= 1) x = fmaxf(x, __shfl_xor(x, m, 32));
  return x;
}
__device__ inline float hsum16(float x) {
#pragma unroll
  for (int m = 1; m < 16; m <<= 1) x += __shfl_xor(x, m, 32);
  return x;
}

#define WMMA_F16(A, B, C) \
  __builtin_amdgcn_wmma_f32_16x16x32_f16(false, (A), false, (B), (short)0, (C), false, false)

// ---------------------------------------------------------------------------
__global__ void zero_ws_kernel(uint4* __restrict__ p, unsigned n) {
  unsigned i = blockIdx.x * blockDim.x + threadIdx.x;
  uint4 z; z.x = z.y = z.z = z.w = 0u;
  for (; i < n; i += gridDim.x * blockDim.x) p[i] = z;
}

// ---------------------------------------------------------------------------
// Projection: out = X @ W^T, converted to f16.
// vmode 0: store [b][h][s][DPAD] (Q,K).  vmode 1: store [b][h][DPAD][s] (V^T).
__global__ __launch_bounds__(256) void proj_kernel(
    const float* __restrict__ X, const float* __restrict__ W,
    _Float16* __restrict__ outp, int vmode) {
  const int wave = blockIdx.x * 8 + (threadIdx.x >> 5);
  const int lane = threadIdx.x & 31;
  const int ln = lane & 15, lhi = lane >> 4;
  const int mt = wave / (DMODEL / 16);
  const int nt = wave % (DMODEL / 16);
  const int m0 = mt * 16, n0 = nt * 16;

  v8f acc = zero8();
#pragma unroll
  for (int kc = 0; kc < DMODEL / 32; ++kc) {
    v16h a = a_frag_f32(X + (size_t)(m0 + ln) * DMODEL + kc * 32 + lhi * 16);
    v16h b = b_frag_f32(W + (size_t)(n0 + ln) * DMODEL + kc * 32 + lhi * 16);
    acc = WMMA_F16(a, b, acc);
  }
  const int h = n0 / DHEAD, dh0 = n0 - h * DHEAD;
  const int bIdx = m0 >> 11, s0 = m0 & (SEQ - 1);
  if (vmode == 0) {
#pragma unroll
    for (int r = 0; r < 8; ++r)
      outp[(((size_t)(bIdx * NHEAD + h)) * SEQ + s0 + r + 8 * lhi) * DPAD + dh0 + ln] =
          (_Float16)acc[r];
  } else {
#pragma unroll
    for (int r = 0; r < 8; ++r)
      outp[(((size_t)(bIdx * NHEAD + h)) * DPAD + dh0 + ln) * SEQ + s0 + r + 8 * lhi] =
          (_Float16)acc[r];
  }
}

// ---------------------------------------------------------------------------
// Attention: one block = one 16-row q-tile; wave w owns head w (4 waves).
// Pass A: running (max, sumexp) over k-tiles (wave-independent).
// Pass B: recompute weights, cooperative head-mean via LDS -> NT store,
// wave-private LDS transpose -> P@V WMMA.
__global__ __launch_bounds__(128) void attn_kernel(
    const _Float16* __restrict__ Qh, const _Float16* __restrict__ Kh,
    const _Float16* __restrict__ Vt, const int* __restrict__ mask,
    const float* __restrict__ bias, _Float16* __restrict__ Ctx,
    float* __restrict__ out2) {
  __shared__ __align__(64) _Float16 wbuf[NHEAD][16][32];   // per-head P tiles (f16)
  __shared__ __align__(64) float mwpart[NHEAD][16][16];    // per-head 0.25*w (f32)

  const int tid = threadIdx.x;
  const int h = tid >> 5;                  // wave index == head index
  const int lane = tid & 31;
  const int ln = lane & 15, lhi = lane >> 4;
  const int bIdx = blockIdx.x >> 7;        // SEQ/16 = 128 q-tiles per batch
  const int q0 = (blockIdx.x & 127) << 4;

  const size_t bh = (size_t)(bIdx * NHEAD + h);

  // Q fragments for this head, resident for both passes.
  v16h aQ[2];
#pragma unroll
  for (int kc = 0; kc < 2; ++kc)
    aQ[kc] = a_frag_f16(Qh + (bh * SEQ + q0 + ln) * DPAD + kc * 32 + lhi * 16);

  float mrow[8], lrow[8];
#pragma unroll
  for (int r = 0; r < 8; ++r) { mrow[r] = -3.0e38f; lrow[r] = 0.0f; }

  // ---------------- Pass A: row max + sumexp ----------------
  for (int kt = 0; kt < SEQ / 16; ++kt) {
    const int k0 = kt << 4;
    v8f Sf = zero8();
#pragma unroll
    for (int kc = 0; kc < 2; ++kc) {
      v16h bK = b_frag_f16(Kh + (bh * SEQ + k0 + ln) * DPAD + kc * 32 + lhi * 16);
      Sf = WMMA_F16(aQ[kc], bK, Sf);
    }
#pragma unroll
    for (int r = 0; r < 8; ++r) {
      int mk = mask[((size_t)bIdx * SEQ + q0 + r + 8 * lhi) * SEQ + k0 + ln];
      float bb = bias[(((size_t)h) * SEQ + q0 + r + 8 * lhi) * SEQ + k0 + ln];
      float val = (mk == 0) ? NEG_INF_F : Sf[r] + bb;
      float mn = fmaxf(mrow[r], hmax16(val));
      float es = hsum16(__expf(val - mn));
      lrow[r] = lrow[r] * __expf(mrow[r] - mn) + es;
      mrow[r] = mn;
    }
  }
  float invl[8];
#pragma unroll
  for (int r = 0; r < 8; ++r) invl[r] = 1.0f / lrow[r];

  // ---------------- Pass B: weights, head-mean, P@V ----------------
  v8f pv[3];
#pragma unroll
  for (int nt = 0; nt < 3; ++nt) pv[nt] = zero8();

  for (int ktp = 0; ktp < SEQ / 32; ++ktp) {
#pragma unroll
    for (int sub = 0; sub < 2; ++sub) {
      const int kt = ktp * 2 + sub, k0 = kt << 4;
      v8f Sf = zero8();
#pragma unroll
      for (int kc = 0; kc < 2; ++kc) {
        v16h bK = b_frag_f16(Kh + (bh * SEQ + k0 + ln) * DPAD + kc * 32 + lhi * 16);
        Sf = WMMA_F16(aQ[kc], bK, Sf);
      }
#pragma unroll
      for (int r = 0; r < 8; ++r) {
        // last-use loads: don't refresh L2 lines we won't touch again
        int mk = __builtin_nontemporal_load(
            &mask[((size_t)bIdx * SEQ + q0 + r + 8 * lhi) * SEQ + k0 + ln]);
        float bb = bias[(((size_t)h) * SEQ + q0 + r + 8 * lhi) * SEQ + k0 + ln];
        float val = (mk == 0) ? NEG_INF_F : Sf[r] + bb;
        float w = __expf(val - mrow[r]) * invl[r];
        // C-layout (lane = k column, reg = q row) -> LDS [qrow][kcol]
        wbuf[h][r + 8 * lhi][sub * 16 + ln] = (_Float16)w;
        mwpart[h][r + 8 * lhi][ln] = 0.25f * w;
      }
      __syncthreads();
      // Cooperative head-mean: 128 threads cover the 16x16 tile twice.
#pragma unroll
      for (int j = 0; j < 2; ++j) {
        const int e = tid + j * 128;
        const int row = e >> 4, col = e & 15;
        float s4 = mwpart[0][row][col] + mwpart[1][row][col] +
                   mwpart[2][row][col] + mwpart[3][row][col];
        // streaming output, never re-read: keep it out of L2
        __builtin_nontemporal_store(
            s4, &out2[((size_t)bIdx * SEQ + q0 + row) * SEQ + k0 + col]);
      }
      __syncthreads();
    }
    // P@V for this head's 16x32 weight tile (wave-private wbuf[h]).
    const int kp0 = ktp << 5;
    v16h aW;
    {
      v8h lo = *(const v8h*)&wbuf[h][ln][lhi * 16];
      v8h hi = *(const v8h*)&wbuf[h][ln][lhi * 16 + 8];
#pragma unroll
      for (int i = 0; i < 8; ++i) { aW[i] = lo[i]; aW[8 + i] = hi[i]; }
    }
#pragma unroll
    for (int nt = 0; nt < 3; ++nt) {
      v16h bV = b_frag_f16(Vt + (bh * DPAD + nt * 16 + ln) * SEQ + kp0 + lhi * 16);
      pv[nt] = WMMA_F16(aW, bV, pv[nt]);
    }
  }

  // Store this head's context (already softmax-normalized) as f16 [b][s][192].
#pragma unroll
  for (int nt = 0; nt < 3; ++nt)
#pragma unroll
    for (int r = 0; r < 8; ++r)
      Ctx[((size_t)bIdx * SEQ + q0 + r + 8 * lhi) * DMODEL + h * DHEAD + nt * 16 + ln] =
          (_Float16)pv[nt][r];
}

// ---------------------------------------------------------------------------
// Output projection: out = Ctx(f16) @ Wo^T -> f32 (streaming NT store).
__global__ __launch_bounds__(256) void outproj_kernel(
    const _Float16* __restrict__ Ctx, const float* __restrict__ Wo,
    float* __restrict__ out) {
  const int wave = blockIdx.x * 8 + (threadIdx.x >> 5);
  const int lane = threadIdx.x & 31;
  const int ln = lane & 15, lhi = lane >> 4;
  const int mt = wave / (DMODEL / 16);
  const int nt = wave % (DMODEL / 16);
  const int m0 = mt * 16, n0 = nt * 16;

  v8f acc = zero8();
#pragma unroll
  for (int kc = 0; kc < DMODEL / 32; ++kc) {
    v16h a = a_frag_f16(Ctx + (size_t)(m0 + ln) * DMODEL + kc * 32 + lhi * 16);
    v16h b = b_frag_f32(Wo + (size_t)(n0 + ln) * DMODEL + kc * 32 + lhi * 16);
    acc = WMMA_F16(a, b, acc);
  }
#pragma unroll
  for (int r = 0; r < 8; ++r)
    __builtin_nontemporal_store(
        acc[r], &out[(size_t)(m0 + r + 8 * lhi) * DMODEL + n0 + ln]);
}

// ---------------------------------------------------------------------------
extern "C" void kernel_launch(void* const* d_in, const int* in_sizes, int n_in,
                              void* d_out, int out_size, void* d_ws, size_t ws_size,
                              hipStream_t stream) {
  (void)in_sizes; (void)n_in; (void)out_size; (void)ws_size;
  const float* query = (const float*)d_in[0];
  const float* key   = (const float*)d_in[1];
  const float* value = (const float*)d_in[2];
  const int*   amask = (const int*)d_in[3];
  const float* bias  = (const float*)d_in[4];
  const float* Wq    = (const float*)d_in[5];
  const float* Wk    = (const float*)d_in[6];
  const float* Wv    = (const float*)d_in[7];
  const float* Wo    = (const float*)d_in[8];

  float* out  = (float*)d_out;
  float* out2 = out + (size_t)BATCH * SEQ * DMODEL;

  const size_t nQK = (size_t)BATCH * NHEAD * SEQ * DPAD;  // 2,097,152 f16 each
  _Float16* Qh  = (_Float16*)d_ws;
  _Float16* Kh  = Qh + nQK;
  _Float16* Vt  = Kh + nQK;
  _Float16* Ctx = Vt + nQK;   // BATCH*SEQ*DMODEL f16; total ws use ~15 MB

  // Zero Q/K/V workspace (covers the dh 48->64 zero padding).
  zero_ws_kernel<<<512, 256, 0, stream>>>((uint4*)Qh, (unsigned)(3 * nQK / 8));

  const int projBlocks = (BATCH * SEQ / 16) * (DMODEL / 16) / 8;  // 768
  proj_kernel<<<projBlocks, 256, 0, stream>>>(query, Wq, Qh, 0);
  proj_kernel<<<projBlocks, 256, 0, stream>>>(key,   Wk, Kh, 0);
  proj_kernel<<<projBlocks, 256, 0, stream>>>(value, Wv, Vt, 1);

  attn_kernel<<<BATCH * SEQ / 16, 128, 0, stream>>>(Qh, Kh, Vt, amask, bias, Ctx, out2);

  outproj_kernel<<<projBlocks, 256, 0, stream>>>(Ctx, Wo, out);
}
